// OnlineMemoryAttention_83442624627083
// MI455X (gfx1250) — matrix operand order repaired and checked by reading, not compiled, not run on hardware
//
#include <hip/hip_runtime.h>
#include <hip/hip_bf16.h>
#include <cmath>

#define T_SEQ   2048
#define DMODEL  1024
#define NHEADS  16
#define DHEAD   64

typedef _Float16 v8h  __attribute__((ext_vector_type(8)));
typedef _Float16 v16h __attribute__((ext_vector_type(16)));
typedef float    v8f  __attribute__((ext_vector_type(8)));
typedef unsigned int u32x4 __attribute__((ext_vector_type(4)));
typedef unsigned int u32x8 __attribute__((ext_vector_type(8)));

__device__ __forceinline__ v16h load2x8(const _Float16* p0, const _Float16* p1) {
  v8h a = *(const v8h*)p0;
  v8h b = *(const v8h*)p1;
  v16h r;
#pragma unroll
  for (int i = 0; i < 8; ++i) { r[i] = a[i]; r[i + 8] = b[i]; }
  return r;
}

__device__ __forceinline__ float sigmoid_f(float x) { return 1.0f / (1.0f + __expf(-x)); }

// ---- Tensor Data Mover: DMA one 2D tile (rows x 32 halves) global -> LDS ----
// D# group0: count=1, lds_addr, 57b global_addr, type=2 ("image")
// D# group1: data_size=2B, tensor_dim0=K, tensor_dim1=rows_remaining (OOB rows
//            read as zero -> free edge-N handling), tile = 32 x 64, dim0 stride=K.
__device__ __forceinline__ void tdm_load_tile(unsigned long long ga, unsigned lds,
                                              unsigned rows_rem, unsigned k) {
  u32x4 g0;
  g0[0] = 1u;                                               // count=1, user mode
  g0[1] = lds;                                              // LDS byte address
  g0[2] = (unsigned)ga;                                     // global_addr[31:0]
  g0[3] = ((unsigned)(ga >> 32) & 0x01FFFFFFu) | (2u << 30); // addr[56:32] | type=2
  u32x8 g1;
  g1[0] = 1u << 16;                                         // data_size=1 (2 bytes)
  g1[1] = (k & 0xFFFFu) << 16;                              // tensor_dim0[15:0]
  g1[2] = (k >> 16) | ((rows_rem & 0xFFFFu) << 16);         // td0[31:16] | td1[15:0]
  g1[3] = (rows_rem >> 16) | (32u << 16);                   // td1[31:16] | tile_dim0=32
  g1[4] = 64u;                                              // tile_dim1=64, tile_dim2=0
  g1[5] = k;                                                // tensor_dim0_stride[31:0]
  g1[6] = 0u;                                               // stride[47:32] | td1_stride lo
  g1[7] = 0u;
  asm volatile("tensor_load_to_lds %0, %1" :: "s"(g0), "s"(g1) : "memory");
}

// ---------------- elementwise conversions ----------------
__global__ void k_cvt_f16(const float* __restrict__ s, _Float16* __restrict__ d, int n) {
  int i = blockIdx.x * blockDim.x + threadIdx.x;
  if (i < n) d[i] = (_Float16)s[i];
}

__global__ void k_xprev_f16(const float* __restrict__ x, _Float16* __restrict__ xp) {
  int i = blockIdx.x * blockDim.x + threadIdx.x;
  if (i < T_SEQ * DMODEL)
    xp[i] = (i < DMODEL) ? (_Float16)0.f : (_Float16)x[i - DMODEL];
}

// ---------------- WMMA GEMM: C[M,N] = A[M,K] * W[N,K]^T (+bias) ----------------
// f16 A/W, f32 accumulate/out. Block tile 64x64, 4 waves, wave w owns cols [16w,16w+16).
// Tiles are staged by the Tensor Data Mover (double-buffered, issued by wave 0,
// completion via TENSORcnt) so VALU slots stay free for v_wmma.
__global__ __launch_bounds__(128)
void k_gemm_nt(const _Float16* __restrict__ A, const _Float16* __restrict__ W,
               const float* __restrict__ bias, float* __restrict__ C,
               int M, int N, int K) {
  __shared__ __align__(32) _Float16 sA[2][64 * 32];
  __shared__ __align__(32) _Float16 sB[2][64 * 32];
  const int tid  = threadIdx.x;
  const int wave = tid >> 5;
  const int lane = tid & 31;
  const int lr   = lane & 15;
  const int hi   = lane >> 4;
  const int m0 = blockIdx.y * 64;
  const int n0 = blockIdx.x * 64;
  const int nk = K >> 5;

  const unsigned long long gaA = (unsigned long long)(A + (size_t)m0 * K);
  const unsigned long long gaB = (unsigned long long)(W + (size_t)n0 * K);
  const unsigned rowsA = (unsigned)(M - m0);
  const unsigned rowsB = (unsigned)(N - n0);
  const unsigned ldsA0 = (unsigned)(unsigned long long)&sA[0][0];
  const unsigned ldsA1 = (unsigned)(unsigned long long)&sA[1][0];
  const unsigned ldsB0 = (unsigned)(unsigned long long)&sB[0][0];
  const unsigned ldsB1 = (unsigned)(unsigned long long)&sB[1][0];

  if (tid < 32) {  // wave 0 drives the TDM
    tdm_load_tile(gaA, ldsA0, rowsA, (unsigned)K);
    tdm_load_tile(gaB, ldsB0, rowsB, (unsigned)K);
  }

  v8f acc[4];
#pragma unroll
  for (int mi = 0; mi < 4; ++mi)
#pragma unroll
    for (int e = 0; e < 8; ++e) acc[mi][e] = 0.f;

  for (int ik = 0; ik < nk; ++ik) {
    if (tid < 32) __builtin_amdgcn_s_wait_tensorcnt(0);
    __syncthreads();                         // tile ik visible to all waves
    if (tid < 32 && (ik + 1) < nk) {         // prefetch tile ik+1 into other buffer
      const unsigned long long koff = (unsigned long long)(ik + 1) * 64ull; // 32 halves
      tdm_load_tile(gaA + koff, (ik & 1) ? ldsA0 : ldsA1, rowsA, (unsigned)K);
      tdm_load_tile(gaB + koff, (ik & 1) ? ldsB0 : ldsB1, rowsB, (unsigned)K);
    }

    const _Float16* a = sA[ik & 1];
    const _Float16* b = sB[ik & 1];
    // fragments: A lane holds row m=lr, K runs {c..c+7, c+16..c+23}, c = hi?8:0
    //            B lane holds col n=lr, same K runs as A
    const int ka = hi ? 8 : 0;
    const _Float16* bp = &b[(wave * 16 + lr) * 32 + ka];
    v16h bfrag = load2x8(bp, bp + 16);
#pragma unroll
    for (int mi = 0; mi < 4; ++mi) {
      const _Float16* ap = &a[(mi * 16 + lr) * 32 + ka];
      v16h afrag = load2x8(ap, ap + 16);
      acc[mi] = __builtin_amdgcn_wmma_f32_16x16x32_f16(
          false, afrag, false, bfrag, (short)0, acc[mi], false, false);
    }
    __syncthreads();                         // reads done before TDM reuses buffer
  }

  const int n = n0 + wave * 16 + lr;
  if (n < N) {
    const float bb = bias ? bias[n] : 0.f;
#pragma unroll
    for (int mi = 0; mi < 4; ++mi) {
#pragma unroll
      for (int r = 0; r < 8; ++r) {
        const int m = m0 + mi * 16 + r + hi * 8;  // C layout: VGPR r -> row r / r+8
        if (m < M) C[(size_t)m * N + n] = acc[mi][r] + bb;
      }
    }
  }
}

// ---------------- split qkv (T,3D) f32 -> Q,K (H,T,64) f16 and V^T (H,64,T) f16 ----
__global__ void k_split_qkv(const float* __restrict__ qkv,
                            _Float16* __restrict__ Qh, _Float16* __restrict__ Kh,
                            _Float16* __restrict__ Vt) {
  const int i = blockIdx.x * blockDim.x + threadIdx.x;
  if (i >= T_SEQ * DMODEL) return;
  const int t = i / DMODEL;
  const int e = i % DMODEL;
  const int h = e >> 6;
  const int d = e & 63;
  const size_t src = (size_t)t * (3 * DMODEL);
  const size_t qk = (size_t)h * T_SEQ * DHEAD + (size_t)t * DHEAD + d;
  Qh[qk] = (_Float16)qkv[src + e];
  Kh[qk] = (_Float16)qkv[src + DMODEL + e];
  Vt[(size_t)h * DHEAD * T_SEQ + (size_t)d * T_SEQ + t] = (_Float16)qkv[src + 2 * DMODEL + e];
}

// ---------------- flash-style causal attention, WMMA f16 ----------------
__global__ __launch_bounds__(128)
void k_flash(const _Float16* __restrict__ Qh, const _Float16* __restrict__ Kh,
             const _Float16* __restrict__ Vt, float* __restrict__ seq_out) {
  __shared__ __align__(32) _Float16 sP[4][16 * 32];  // per-wave P scratch
  const int lane = threadIdx.x & 31;
  const int wave = threadIdx.x >> 5;
  const int lr = lane & 15;
  const int hi = lane >> 4;
  const int h  = blockIdx.y;
  const int q0 = blockIdx.x * 64 + wave * 16;
  const size_t headQK = (size_t)h * T_SEQ * DHEAD;
  const size_t headV  = (size_t)h * DHEAD * T_SEQ;

  // Q fragments (16 rows x dh=64 => two K=32 A-frags), loaded once
  v16h qf[2];
  {
    const _Float16* qrow = Qh + headQK + (size_t)(q0 + lr) * DHEAD;
    const int c = hi ? 16 : 0;
    qf[0] = *(const v16h*)(qrow + c);
    qf[1] = *(const v16h*)(qrow + 32 + c);
  }

  v8f o[4];
#pragma unroll
  for (int nf = 0; nf < 4; ++nf)
#pragma unroll
    for (int e = 0; e < 8; ++e) o[nf][e] = 0.f;

  float mrow[8], lrow[8];
#pragma unroll
  for (int r = 0; r < 8; ++r) { mrow[r] = -INFINITY; lrow[r] = 0.f; }

  const float scale = 0.125f;  // 64^-0.5

  for (int j0 = 0; j0 <= q0 + 15; j0 += 32) {
    // S(16x32) = Q(16x64) * K^T(64x32): 4 WMMAs
    v8f s0, s1;
#pragma unroll
    for (int e = 0; e < 8; ++e) { s0[e] = 0.f; s1[e] = 0.f; }
    {
      const _Float16* kp0 = Kh + headQK + (size_t)(j0 + lr) * DHEAD + (hi ? 16 : 0);
      const _Float16* kp1 = Kh + headQK + (size_t)(j0 + 16 + lr) * DHEAD + (hi ? 16 : 0);
      v16h b00 = *(const v16h*)kp0;
      v16h b01 = *(const v16h*)(kp0 + 32);
      v16h b10 = *(const v16h*)kp1;
      v16h b11 = *(const v16h*)(kp1 + 32);
      if (j0 + 32 <= q0 + 15) {  // prefetch next K rows
        __builtin_prefetch(kp0 + 32 * DHEAD, 0, 1);
        __builtin_prefetch(kp1 + 32 * DHEAD, 0, 1);
      }
      s0 = __builtin_amdgcn_wmma_f32_16x16x32_f16(false, qf[0], false, b00, (short)0, s0, false, false);
      s0 = __builtin_amdgcn_wmma_f32_16x16x32_f16(false, qf[1], false, b01, (short)0, s0, false, false);
      s1 = __builtin_amdgcn_wmma_f32_16x16x32_f16(false, qf[0], false, b10, (short)0, s1, false, false);
      s1 = __builtin_amdgcn_wmma_f32_16x16x32_f16(false, qf[1], false, b11, (short)0, s1, false, false);
    }

    // causal mask + online softmax (rows live in 16-lane halves -> shfl_xor 1..8)
    float ps0[8], ps1[8];
#pragma unroll
    for (int r = 0; r < 8; ++r) {
      const int t = q0 + r + hi * 8;
      float a = s0[r] * scale; if (j0 + lr > t)      a = -INFINITY;
      float b = s1[r] * scale; if (j0 + 16 + lr > t) b = -INFINITY;
      float vmax = fmaxf(a, b);
#pragma unroll
      for (int msk = 1; msk <= 8; msk <<= 1)
        vmax = fmaxf(vmax, __shfl_xor(vmax, msk, 32));
      const float mnew  = fmaxf(mrow[r], vmax);
      const float alpha = __expf(mrow[r] - mnew);
      const float p0 = __expf(a - mnew);
      const float p1 = __expf(b - mnew);
      float psum = p0 + p1;
#pragma unroll
      for (int msk = 1; msk <= 8; msk <<= 1)
        psum += __shfl_xor(psum, msk, 32);
      lrow[r] = lrow[r] * alpha + psum;
      mrow[r] = mnew;
      ps0[r] = p0; ps1[r] = p1;
#pragma unroll
      for (int nf = 0; nf < 4; ++nf) o[nf][r] *= alpha;
    }

    // re-fragment P via per-wave LDS: C layout -> A layout
    {
      _Float16* pb = &sP[wave][0];
#pragma unroll
      for (int r = 0; r < 8; ++r) {
        const int m = r + hi * 8;
        pb[m * 32 + lr]      = (_Float16)ps0[r];
        pb[m * 32 + 16 + lr] = (_Float16)ps1[r];
      }
    }
    asm volatile("s_wait_dscnt 0" ::: "memory");  // intra-wave LDS RAW
    v16h pf;
    {
      const _Float16* pr = &sP[wave][lr * 32 + (hi ? 16 : 0)];
      pf = *(const v16h*)pr;
    }
    // O(16x64) += P(16x32) * V(32x64): 4 WMMAs, V stored transposed (dh,T)
#pragma unroll
    for (int nf = 0; nf < 4; ++nf) {
      const _Float16* vp = Vt + headV + (size_t)(nf * 16 + lr) * T_SEQ + j0 + (hi ? 16 : 0);
      v16h vf = *(const v16h*)vp;
      o[nf] = __builtin_amdgcn_wmma_f32_16x16x32_f16(false, pf, false, vf, (short)0, o[nf], false, false);
    }
  }

#pragma unroll
  for (int r = 0; r < 8; ++r) {
    const int t = q0 + r + hi * 8;
    const float inv = 1.0f / lrow[r];
#pragma unroll
    for (int nf = 0; nf < 4; ++nf)
      seq_out[(size_t)t * DMODEL + h * DHEAD + nf * 16 + lr] = o[nf][r] * inv;
  }
}

// ---------------- Pluecker lines: exterior + normalize (+ J6 for write) ----------
__device__ __forceinline__ void exterior6(const float* p1, const float* p2, float* o6) {
  o6[0] = p1[0]*p2[1] - p1[1]*p2[0];
  o6[1] = p1[0]*p2[2] - p1[2]*p2[0];
  o6[2] = p1[0]*p2[3] - p1[3]*p2[0];
  o6[3] = p1[1]*p2[2] - p1[2]*p2[1];
  o6[4] = p1[1]*p2[3] - p1[3]*p2[1];
  o6[5] = p1[2]*p2[3] - p1[3]*p2[2];
  float ss = 0.f;
#pragma unroll
  for (int a = 0; a < 6; ++a) ss += o6[a] * o6[a];
  ss = fmaxf(ss, 1e-24f);
  const float inv = 1.0f / sqrtf(ss);
#pragma unroll
  for (int a = 0; a < 6; ++a) o6[a] *= inv;
}

__global__ void k_pluecker(const float* __restrict__ w1, const float* __restrict__ w2,
                           const float* __restrict__ r1, const float* __restrict__ r2,
                           float* __restrict__ rdl, float* __restrict__ jwr) {
  const int i = blockIdx.x * blockDim.x + threadIdx.x;  // i = t*NHEADS + h
  if (i >= T_SEQ * NHEADS) return;
  const int t = i / NHEADS, h = i % NHEADS;
  const size_t base = (size_t)t * 64 + h * 4;
  float a[4], b[4], L[6];
#pragma unroll
  for (int c = 0; c < 4; ++c) { a[c] = w1[base + c]; b[c] = w2[base + c]; }
  exterior6(a, b, L);
  float* jw = jwr + (size_t)i * 6;               // write_lines * J6
  jw[0] =  L[5]; jw[1] = -L[4]; jw[2] =  L[3];
  jw[3] =  L[2]; jw[4] = -L[1]; jw[5] =  L[0];
#pragma unroll
  for (int c = 0; c < 4; ++c) { a[c] = r1[base + c]; b[c] = r2[base + c]; }
  exterior6(a, b, L);
  float* rl = rdl + (size_t)i * 6;
#pragma unroll
  for (int c = 0; c < 6; ++c) rl[c] = L[c];
}

// ---------------- strict-causal incidence^2 score, LDS-staged j_write ----------
__global__ __launch_bounds__(256)
void k_memscore(const float* __restrict__ rdl, const float* __restrict__ jwr,
                float* __restrict__ score) {
  __shared__ float sj[256 * 6];
  const int h = blockIdx.y;
  const int t = blockIdx.x * 256 + threadIdx.x;
  float rv[6];
#pragma unroll
  for (int a = 0; a < 6; ++a) rv[a] = rdl[((size_t)t * NHEADS + h) * 6 + a];
  float acc = 0.f;
  const int smax = blockIdx.x * 256 + 256;
  for (int s0 = 0; s0 < smax; s0 += 256) {
    __syncthreads();
    {
      const int s = s0 + threadIdx.x;
      const float* src = jwr + ((size_t)s * NHEADS + h) * 6;
#pragma unroll
      for (int a = 0; a < 6; ++a) sj[threadIdx.x * 6 + a] = src[a];
    }
    __syncthreads();
    const int send = min(256, t - s0);  // strict s < t
    for (int ss = 0; ss < send; ++ss) {
      const float* j = &sj[ss * 6];
      float d = rv[0]*j[0] + rv[1]*j[1] + rv[2]*j[2]
              + rv[3]*j[3] + rv[4]*j[4] + rv[5]*j[5];
      acc += d * d;
    }
  }
  score[(size_t)t * NHEADS + h] = acc;
}

// ---------------- gate + combine -> f16 for final projection ----------------
__global__ __launch_bounds__(256)
void k_combine(const float* __restrict__ seq, const float* __restrict__ mval,
               const float* __restrict__ score, const float* __restrict__ gpre,
               const float* __restrict__ mem_scale, _Float16* __restrict__ comb_h) {
  const int t = blockIdx.x;
  __shared__ float gsh;
  if (threadIdx.x == 0) {
    float s = 0.f;
#pragma unroll
    for (int h = 0; h < NHEADS; ++h)
      s += sigmoid_f(score[(size_t)t * NHEADS + h] * mem_scale[h]) *
           sigmoid_f(gpre[(size_t)t * NHEADS + h]);
    gsh = s * (1.0f / NHEADS);
  }
  __syncthreads();
  const float g = gsh;
  for (int d = threadIdx.x; d < DMODEL; d += 256) {
    const size_t idx = (size_t)t * DMODEL + d;
    comb_h[idx] = (_Float16)(seq[idx] + g * mval[idx]);
  }
}

// ---------------- host launch ----------------
extern "C" void kernel_launch(void* const* d_in, const int* in_sizes, int n_in,
                              void* d_out, int out_size, void* d_ws, size_t ws_size,
                              hipStream_t stream) {
  (void)in_sizes; (void)n_in; (void)out_size; (void)ws_size;
  const float* x      = (const float*)d_in[0];
  const float* qkv_w  = (const float*)d_in[1];
  const float* qkv_b  = (const float*)d_in[2];
  const float* w1w    = (const float*)d_in[3];
  const float* w2w    = (const float*)d_in[4];
  const float* w1r    = (const float*)d_in[5];
  const float* w2r    = (const float*)d_in[6];
  const float* mvw    = (const float*)d_in[7];
  const float* mvb    = (const float*)d_in[8];
  const float* gw     = (const float*)d_in[9];
  const float* gb     = (const float*)d_in[10];
  const float* mscale = (const float*)d_in[11];
  const float* ow     = (const float*)d_in[12];
  const float* ob     = (const float*)d_in[13];

  char* ws = (char*)d_ws;
  size_t off = 0;
  auto alloc = [&](size_t bytes) -> void* {
    off = (off + 255) & ~(size_t)255;
    void* p = ws + off;
    off += bytes;
    return p;
  };

  const int T = T_SEQ, D = DMODEL, H = NHEADS;
  _Float16* x_h    = (_Float16*)alloc((size_t)T * D * 2);
  _Float16* xp_h   = (_Float16*)alloc((size_t)T * D * 2);
  _Float16* qkvw_h = (_Float16*)alloc((size_t)3 * D * D * 2);
  _Float16* w1w_h  = (_Float16*)alloc((size_t)4 * H * D * 2);
  _Float16* w2w_h  = (_Float16*)alloc((size_t)4 * H * D * 2);
  _Float16* w1r_h  = (_Float16*)alloc((size_t)4 * H * D * 2);
  _Float16* w2r_h  = (_Float16*)alloc((size_t)4 * H * D * 2);
  _Float16* mvw_h  = (_Float16*)alloc((size_t)D * D * 2);
  _Float16* gw_h   = (_Float16*)alloc((size_t)H * D * 2);
  _Float16* ow_h   = (_Float16*)alloc((size_t)D * D * 2);
  _Float16* comb_h = (_Float16*)alloc((size_t)T * D * 2);
  _Float16* Qh     = (_Float16*)alloc((size_t)H * T * DHEAD * 2);
  _Float16* Kh     = (_Float16*)alloc((size_t)H * T * DHEAD * 2);
  _Float16* Vt     = (_Float16*)alloc((size_t)H * DHEAD * T * 2);
  float* qkv_out = (float*)alloc((size_t)T * 3 * D * 4);
  float* w1o     = (float*)alloc((size_t)T * 64 * 4);
  float* w2o     = (float*)alloc((size_t)T * 64 * 4);
  float* r1o     = (float*)alloc((size_t)T * 64 * 4);
  float* r2o     = (float*)alloc((size_t)T * 64 * 4);
  float* rdl     = (float*)alloc((size_t)T * H * 6 * 4);
  float* jwr     = (float*)alloc((size_t)T * H * 6 * 4);
  float* score   = (float*)alloc((size_t)T * H * 4);
  float* mval    = (float*)alloc((size_t)T * D * 4);
  float* gpre    = (float*)alloc((size_t)T * H * 4);
  float* seq     = (float*)alloc((size_t)T * D * 4);

  auto cvt = [&](const float* s, _Float16* d, int n) {
    k_cvt_f16<<<dim3((n + 255) / 256), dim3(256), 0, stream>>>(s, d, n);
  };
  cvt(x, x_h, T * D);
  k_xprev_f16<<<dim3((T * D + 255) / 256), dim3(256), 0, stream>>>(x, xp_h);
  cvt(qkv_w, qkvw_h, 3 * D * D);
  cvt(w1w, w1w_h, 4 * H * D);
  cvt(w2w, w2w_h, 4 * H * D);
  cvt(w1r, w1r_h, 4 * H * D);
  cvt(w2r, w2r_h, 4 * H * D);
  cvt(mvw, mvw_h, D * D);
  cvt(gw, gw_h, H * D);
  cvt(ow, ow_h, D * D);

  auto gemm = [&](const _Float16* A, const _Float16* W, const float* bias,
                  float* C, int M, int N, int K) {
    dim3 g((N + 63) / 64, M / 64);
    k_gemm_nt<<<g, dim3(128), 0, stream>>>(A, W, bias, C, M, N, K);
  };
  gemm(x_h,  qkvw_h, qkv_b,   qkv_out, T, 3 * D, D);
  gemm(xp_h, w1w_h,  nullptr, w1o,     T, 64,    D);
  gemm(x_h,  w2w_h,  nullptr, w2o,     T, 64,    D);
  gemm(x_h,  w1r_h,  nullptr, r1o,     T, 64,    D);
  gemm(x_h,  w2r_h,  nullptr, r2o,     T, 64,    D);
  gemm(x_h,  mvw_h,  mvb,     mval,    T, D,     D);
  gemm(x_h,  gw_h,   gb,      gpre,    T, H,     D);

  k_split_qkv<<<dim3((T * D + 255) / 256), dim3(256), 0, stream>>>(qkv_out, Qh, Kh, Vt);
  k_flash<<<dim3(T / 64, H), dim3(128), 0, stream>>>(Qh, Kh, Vt, seq);

  k_pluecker<<<dim3((T * H + 255) / 256), dim3(256), 0, stream>>>(w1o, w2o, r1o, r2o, rdl, jwr);
  k_memscore<<<dim3(T / 256, H), dim3(256), 0, stream>>>(rdl, jwr, score);
  k_combine<<<dim3(T), dim3(256), 0, stream>>>(seq, mval, score, gpre, mscale, comb_h);

  gemm(comb_h, ow_h, ob, (float*)d_out, T, D, D);
}
